// GraphConvNN_16578573763457
// MI455X (gfx1250) — matrix operand order repaired and checked by reading, not compiled, not run on hardware
//
#include <hip/hip_runtime.h>

#define B_ 8
#define N_ 10000
#define F_ 64
#define H_ 64
#define E_ 160000

typedef __bf16 v16bf __attribute__((ext_vector_type(16)));
typedef float  v8f   __attribute__((ext_vector_type(8)));
typedef unsigned short u16;
typedef unsigned short us8  __attribute__((ext_vector_type(8)));   // 16 bytes
typedef unsigned short us16 __attribute__((ext_vector_type(16)));  // 32 bytes

union BfPack { us16 u; v16bf b; };

// round-to-nearest-even float -> bf16 (bit pattern)
__device__ __forceinline__ u16 f2bf(float f) {
  unsigned u = __float_as_uint(f);
  u += 0x7FFFu + ((u >> 16) & 1u);
  return (u16)(u >> 16);
}

// CDNA5 hardware tanh (single TRANS32 op; confirmed lowering to v_tanh_f32)
__device__ __forceinline__ float fast_tanh(float x) {
#if __has_builtin(__builtin_amdgcn_tanhf)
  return __builtin_amdgcn_tanhf(x);
#else
  float r;
  asm volatile("v_tanh_f32 %0, %1\n\tv_nop" : "=v"(r) : "v"(x));
  return r;
#endif
}

__device__ __forceinline__ v16bf combine(us8 lo, us8 hi) {
  BfPack p;
#pragma unroll
  for (int i = 0; i < 8; ++i) { p.u[i] = lo[i]; p.u[i + 8] = hi[i]; }
  return p.b;
}

// A-matrix (16x32 bf16) lane slice per CDNA5 layout:
// lane l: row = l&15; half = l>>4; holds K = kc+8*half+[0..8) and kc+8*half+16+[0..8)
__device__ __forceinline__ v16bf load_a(const u16* rowp, int kc, int half) {
  const int kb = kc + 8 * half;
  us8 lo = *(const us8*)(rowp + kb);
  us8 hi = *(const us8*)(rowp + kb + 16);
  return combine(lo, hi);
}

// B-matrix (32x16 bf16) lane slice: lane l: col = l&15; holds K = kc+8*(l>>4)+[0..8) and kc+8*(l>>4)+16+[0..8)
// colp points at WT[col*Kstride] (weights transposed: contiguous in K)
__device__ __forceinline__ v16bf load_b(const u16* colp, int kc, int half) {
  const u16* p = colp + kc + 8 * half;
  us8 lo = *(const us8*)(p);
  us8 hi = *(const us8*)(p + 16);
  return combine(lo, hi);
}

__device__ __forceinline__ v8f wmma_bf16(v16bf a, v16bf b, v8f c) {
  return __builtin_amdgcn_wmma_f32_16x16x32_bf16(false, a, false, b, (short)0, c,
                                                 false, false);
}

__device__ __forceinline__ void ds_fence() {
  asm volatile("s_wait_dscnt 0" ::: "memory");
}

// ---------------------------------------------------------------- utility
__global__ void k_zero(float* __restrict__ p, int n) {
  int i = blockIdx.x * 256 + threadIdx.x;
  if (i < n) p[i] = 0.0f;
}

__global__ void k_x2bf(const float* __restrict__ x, u16* __restrict__ xb, int n) {
  int i = blockIdx.x * 256 + threadIdx.x;
  if (i < n) xb[i] = f2bf(x[i]);
}

__global__ void k_counts(const int* __restrict__ node_idx, float* __restrict__ cnt) {
  int e = blockIdx.x * 256 + threadIdx.x;
  if (e < E_) atomicAdd(cnt + node_idx[e], 1.0f);
}

// One-shot: transpose + convert all weights to bf16 (stays L2-hot; 40 KB).
// flat index space: [0,4096) W1T, [4096,8192) W2T, [8192,16384) U1T, [16384,20480) U2T
__global__ void k_wprep(const float* __restrict__ W1, const float* __restrict__ W2,
                        const float* __restrict__ U1, const float* __restrict__ U2,
                        u16* __restrict__ wW1T, u16* __restrict__ wW2T,
                        u16* __restrict__ wU1T, u16* __restrict__ wU2T) {
  int i = blockIdx.x * 256 + threadIdx.x;
  if (i < 4096) {
    int n = i >> 6, k = i & 63;
    wW1T[n * 64 + k] = f2bf(W1[k * 64 + n]);
  } else if (i < 8192) {
    int j = i - 4096, n = j >> 6, k = j & 63;
    wW2T[n * 64 + k] = f2bf(W2[k * 64 + n]);
  } else if (i < 16384) {
    int j = i - 8192, n = j >> 7, k = j & 127;
    wU1T[n * 128 + k] = f2bf(U1[k * 64 + n]);
  } else if (i < 20480) {
    int j = i - 16384, n = j >> 6, k = j & 63;
    wU2T[n * 64 + k] = f2bf(U2[k * 64 + n]);
  }
}

// ------------------------------------------------- message + scatter-sum
// grid.x = E/16 edge tiles, 256 threads = 8 waves, wave w handles batch b=w.
__global__ __launch_bounds__(256)
void k_message(const u16* __restrict__ xbf,
               const int* __restrict__ nbr_idx,
               const int* __restrict__ node_idx,
               const u16* __restrict__ wW1T, const float* __restrict__ b1,
               const u16* __restrict__ wW2T, const float* __restrict__ b2,
               float* __restrict__ sums) {
  __shared__ __attribute__((aligned(16))) u16 sW1[H_ * F_];     // W1^T [H][F]
  __shared__ __attribute__((aligned(16))) u16 sW2[H_ * H_];     // W2^T [H][H]
  __shared__ __attribute__((aligned(16))) u16 sHid[8][16 * 72]; // per-wave staging

  const int t = threadIdx.x;
  {  // straight b128 copy of pre-converted weights into LDS (512 us8 each)
    const us8* g1 = (const us8*)wW1T;  us8* s1 = (us8*)sW1;
    const us8* g2 = (const us8*)wW2T;  us8* s2 = (us8*)sW2;
#pragma unroll
    for (int i = 0; i < 2; ++i) {
      s1[t + i * 256] = g1[t + i * 256];
      s2[t + i * 256] = g2[t + i * 256];
    }
  }
  __syncthreads();

  const int lane = t & 31;
  const int b    = t >> 5;       // wave id == batch
  const int row  = lane & 15;    // A row / D column within tile
  const int half = lane >> 4;
  const int tile = blockIdx.x;

  // gather neighbour feature row (bf16, L2 resident)
  const int e = tile * 16 + row;
  const int j = nbr_idx[e];
  const u16* xrow = xbf + ((size_t)b * N_ + (size_t)j) * F_;
  v16bf A0 = load_a(xrow, 0, half);
  v16bf A1 = load_a(xrow, 32, half);

  u16* hid = &sHid[b][0];
#pragma unroll
  for (int nt = 0; nt < 4; ++nt) {              // layer 1: [16,64]x[64,64]
    const int hcol = nt * 16 + row;
    const float bias = b1[hcol];
    v8f acc;
#pragma unroll
    for (int r = 0; r < 8; ++r) acc[r] = bias;
    const u16* wcol = sW1 + hcol * F_;
    acc = wmma_bf16(A0, load_b(wcol, 0, half), acc);
    acc = wmma_bf16(A1, load_b(wcol, 32, half), acc);
#pragma unroll
    for (int r = 0; r < 8; ++r)
      hid[(8 * half + r) * 72 + hcol] = f2bf(fast_tanh(acc[r]));
  }
  ds_fence();

  const u16* hrow = hid + row * 72;
  v16bf G0 = load_a(hrow, 0, half);
  v16bf G1 = load_a(hrow, 32, half);

  const int4* np = (const int4*)(node_idx + tile * 16 + 8 * half);
  int4 n0 = np[0], n1 = np[1];
  int nidx[8] = {n0.x, n0.y, n0.z, n0.w, n1.x, n1.y, n1.z, n1.w};

#pragma unroll
  for (int nt = 0; nt < 4; ++nt) {              // layer 2 + scatter
    const int hcol = nt * 16 + row;
    const float bias = b2[hcol];
    v8f acc;
#pragma unroll
    for (int r = 0; r < 8; ++r) acc[r] = bias;
    const u16* wcol = sW2 + hcol * H_;
    acc = wmma_bf16(G0, load_b(wcol, 0, half), acc);
    acc = wmma_bf16(G1, load_b(wcol, 32, half), acc);
#pragma unroll
    for (int r = 0; r < 8; ++r)
      atomicAdd(sums + ((size_t)nidx[r] * B_ + b) * H_ + hcol, fast_tanh(acc[r]));
  }
}

// ------------------------------------------------- mean + concat + update
// grid.x = N/16 node tiles, 8 waves = 8 batches.
__global__ __launch_bounds__(256)
void k_update(const u16* __restrict__ xbf,
              const float* __restrict__ sums, const float* __restrict__ cnts,
              const u16* __restrict__ wU1T, const float* __restrict__ ub1,
              const u16* __restrict__ wU2T, const float* __restrict__ ub2,
              float* __restrict__ out) {
  __shared__ __attribute__((aligned(16))) u16 sU1[H_ * 128];      // U1^T [64][128]
  __shared__ __attribute__((aligned(16))) u16 sU2[H_ * H_];       // U2^T [64][64]
  __shared__ __attribute__((aligned(16))) u16 sConc[8][16 * 136]; // concat rows
  __shared__ __attribute__((aligned(16))) u16 sHid[8][16 * 72];

  const int t = threadIdx.x;
  {  // straight b128 copy of pre-converted weights (1024 + 512 us8)
    const us8* g1 = (const us8*)wU1T;  us8* s1 = (us8*)sU1;
    const us8* g2 = (const us8*)wU2T;  us8* s2 = (us8*)sU2;
#pragma unroll
    for (int i = 0; i < 4; ++i) s1[t + i * 256] = g1[t + i * 256];
#pragma unroll
    for (int i = 0; i < 2; ++i) s2[t + i * 256] = g2[t + i * 256];
  }
  __syncthreads();

  const int lane = t & 31;
  const int b    = t >> 5;
  const int row  = lane & 15;
  const int half = lane >> 4;
  const int tile = blockIdx.x;
  const int node = tile * 16 + row;

  u16* conc = &sConc[b][0];
  if (half == 0) {  // cols [0,64): copy x row (bf16)
    const us8* src = (const us8*)(xbf + ((size_t)b * N_ + node) * F_);
#pragma unroll
    for (int i = 0; i < 8; ++i)
      *(us8*)(conc + row * 136 + i * 8) = src[i];
  } else {          // cols [64,128): agg = sums / max(cnt,1)
    const float inv = 1.0f / fmaxf(cnts[node], 1.0f);
    const float4* sp = (const float4*)(sums + ((size_t)node * B_ + b) * H_);
#pragma unroll
    for (int i = 0; i < 8; ++i) {
      float4 s0 = sp[2 * i], s1 = sp[2 * i + 1];
      us8 v;
      v[0] = f2bf(s0.x * inv); v[1] = f2bf(s0.y * inv);
      v[2] = f2bf(s0.z * inv); v[3] = f2bf(s0.w * inv);
      v[4] = f2bf(s1.x * inv); v[5] = f2bf(s1.y * inv);
      v[6] = f2bf(s1.z * inv); v[7] = f2bf(s1.w * inv);
      *(us8*)(conc + row * 136 + 64 + i * 8) = v;
    }
  }
  ds_fence();

  const u16* crow = conc + row * 136;
  v16bf A0 = load_a(crow, 0, half);
  v16bf A1 = load_a(crow, 32, half);
  v16bf A2 = load_a(crow, 64, half);
  v16bf A3 = load_a(crow, 96, half);

  u16* hid = &sHid[b][0];
#pragma unroll
  for (int nt = 0; nt < 4; ++nt) {              // layer 1: [16,128]x[128,64]
    const int hcol = nt * 16 + row;
    const float bias = ub1[hcol];
    v8f acc;
#pragma unroll
    for (int r = 0; r < 8; ++r) acc[r] = bias;
    const u16* wcol = sU1 + hcol * 128;
    acc = wmma_bf16(A0, load_b(wcol, 0, half), acc);
    acc = wmma_bf16(A1, load_b(wcol, 32, half), acc);
    acc = wmma_bf16(A2, load_b(wcol, 64, half), acc);
    acc = wmma_bf16(A3, load_b(wcol, 96, half), acc);
#pragma unroll
    for (int r = 0; r < 8; ++r)
      hid[(8 * half + r) * 72 + hcol] = f2bf(fast_tanh(acc[r]));
  }
  ds_fence();

  const u16* hrow = hid + row * 72;
  v16bf G0 = load_a(hrow, 0, half);
  v16bf G1 = load_a(hrow, 32, half);
#pragma unroll
  for (int nt = 0; nt < 4; ++nt) {              // layer 2 + store
    const int hcol = nt * 16 + row;
    const float bias = ub2[hcol];
    v8f acc;
#pragma unroll
    for (int r = 0; r < 8; ++r) acc[r] = bias;
    const u16* wcol = sU2 + hcol * H_;
    acc = wmma_bf16(G0, load_b(wcol, 0, half), acc);
    acc = wmma_bf16(G1, load_b(wcol, 32, half), acc);
#pragma unroll
    for (int r = 0; r < 8; ++r) {
      const int m = 8 * half + r;
      out[((size_t)b * N_ + (tile * 16 + m)) * H_ + hcol] = fast_tanh(acc[r]);
    }
  }
}

extern "C" void kernel_launch(void* const* d_in, const int* in_sizes, int n_in,
                              void* d_out, int out_size, void* d_ws, size_t ws_size,
                              hipStream_t stream) {
  (void)in_sizes; (void)n_in; (void)out_size; (void)ws_size;
  const float* x        = (const float*)d_in[0];
  const int*   node_idx = (const int*)d_in[1];
  const int*   nbr_idx  = (const int*)d_in[2];
  const float* W1  = (const float*)d_in[3];
  const float* b1  = (const float*)d_in[4];
  const float* W2  = (const float*)d_in[5];
  const float* b2  = (const float*)d_in[6];
  const float* U1  = (const float*)d_in[7];
  const float* ub1 = (const float*)d_in[8];
  const float* U2  = (const float*)d_in[9];
  const float* ub2 = (const float*)d_in[10];
  float* out = (float*)d_out;

  char* ws = (char*)d_ws;
  u16*   xbf  = (u16*)ws;                                   // 10,240,000 B
  float* sums = (float*)(ws + 10240000);                    // 20,480,000 B
  float* cnts = (float*)(ws + 10240000 + 20480000);         //     40,000 B
  u16*   wW1T = (u16*)(ws + 30760192);                      //      8,192 B
  u16*   wW2T = (u16*)(ws + 30760192 + 8192);               //      8,192 B
  u16*   wU1T = (u16*)(ws + 30760192 + 16384);              //     16,384 B
  u16*   wU2T = (u16*)(ws + 30760192 + 32768);              //      8,192 B

  const int zn = N_ * B_ * H_ + N_;  // sums + counts, contiguous
  k_zero<<<(zn + 255) / 256, 256, 0, stream>>>(sums, zn);

  const int xn = B_ * N_ * F_;
  k_x2bf<<<(xn + 255) / 256, 256, 0, stream>>>(x, xbf, xn);

  k_wprep<<<80, 256, 0, stream>>>(W1, W2, U1, U2, wW1T, wW2T, wU1T, wU2T);

  k_counts<<<(E_ + 255) / 256, 256, 0, stream>>>(node_idx, cnts);

  k_message<<<E_ / 16, 256, 0, stream>>>(xbf, nbr_idx, node_idx,
                                         wW1T, b1, wW2T, b2, sums);

  k_update<<<N_ / 16, 256, 0, stream>>>(xbf, sums, cnts,
                                        wU1T, ub1, wU2T, ub2, out);
}
